// TriangleMultiplication_23476291239986
// MI455X (gfx1250) — compile-verified, matrix operand order checked by reading, not run on hardware
//
#include <hip/hip_runtime.h>

// ---------------------------------------------------------------------------
// Triangle multiplication (outgoing), N=512, C=128, fp32 in/out.
// All matmuls via v_wmma_f32_16x16x32_f16 (gfx1250 / CDNA5, wave32).
// k2 uses GLOBAL_LOAD_ASYNC_TO_LDS_B128 double-buffering when available.
// ---------------------------------------------------------------------------

constexpr int N_ = 512;
constexpr int C_ = 128;

typedef __attribute__((ext_vector_type(16))) _Float16 v16h;
typedef __attribute__((ext_vector_type(8)))  float    v8f;
typedef __attribute__((ext_vector_type(4)))  float    v4f;
typedef int v4i __attribute__((vector_size(16)));   // matches builtin param type

union HFrag { v16h v; _Float16 h[16]; float4 q[2]; };
union F8    { v8f  v; v4f p[2]; float f[8]; };
union H8    { _Float16 h[8]; float4 q; };

#if defined(__HIP_DEVICE_COMPILE__) && \
    __has_builtin(__builtin_amdgcn_global_load_async_to_lds_b128) && \
    __has_builtin(__builtin_amdgcn_s_wait_asynccnt)
#define HAVE_ASYNC_LDS 1
#else
#define HAVE_ASYNC_LDS 0
#endif

// 16B global -> LDS copy. Async (ASYNCcnt-tracked, no VGPR staging) on CDNA5.
__device__ __forceinline__ void cp16_to_lds(_Float16* dst, const _Float16* src) {
#if HAVE_ASYNC_LDS
  // Builtin signature (from clang diagnostic): (v4i AS1*, v4i AS3*, Ii, Ii).
  // Flat address of __shared__ has the LDS byte offset in its low 32 bits
  // (ISA 10.2, LDS aperture).
  typedef __attribute__((address_space(1))) v4i* gp_t;
  typedef __attribute__((address_space(3))) v4i* lp_t;
  gp_t g = (gp_t)(size_t)src;
  lp_t l = (lp_t)(unsigned int)(size_t)dst;
  __builtin_amdgcn_global_load_async_to_lds_b128(g, l, 0, 0);
#else
  *reinterpret_cast<float4*>(dst) = *reinterpret_cast<const float4*>(src);
#endif
}

__device__ __forceinline__ void wait_cp_async() {
#if HAVE_ASYNC_LDS
  __builtin_amdgcn_s_wait_asynccnt(0);
#endif
}

// Load a 16x16x32 f16 WMMA fragment (A or B role) for this lane from a
// row-major array: rowp already points at this lane's row (row = base + (l&15));
// k0 must include the lane's k-select: k0 = kchunk*32 + 8*(lane>>4).
// Per ISA layout: lane holds halfs {k0..k0+7, k0+16..k0+23}.
__device__ __forceinline__ v16h frag_ld(const _Float16* rowp, int k0) {
  HFrag f;
  f.q[0] = *reinterpret_cast<const float4*>(rowp + k0);
  f.q[1] = *reinterpret_cast<const float4*>(rowp + k0 + 16);
  return f.v;
}

__device__ __forceinline__ float sigmoidf_(float x) {
  return 1.0f / (1.0f + __expf(-x));
}

#define WMMA_F16(A, B, Cacc) \
  __builtin_amdgcn_wmma_f32_16x16x32_f16(false, (A), false, (B), (short)0, (Cacc), false, false)

// ---------------------------------------------------------------------------
// k0: fp32 -> f16 weight conversion (one 128x128 matrix per launch)
// ---------------------------------------------------------------------------
__global__ __launch_bounds__(256) void k0_cvt_w(const float* __restrict__ s,
                                                _Float16* __restrict__ d) {
  int i = blockIdx.x * 256 + threadIdx.x;
  d[i] = (_Float16)s[i];
}

// ---------------------------------------------------------------------------
// k1: LayerNorm + 5 projections (left/right proj*gate, final gate).
//   outputs: leftO/rightO f16 in [c][p] layout, gO fp32 in [p][c] layout.
//   block = 256 threads (8 waves), 64 positions per block.
//   wave w: position-subtile pt = w&3 (16 pos), out-channel half h = w>>2.
// ---------------------------------------------------------------------------
__global__ __launch_bounds__(256) void k1_prep(
    const float* __restrict__ act, const float* __restrict__ mask,
    const float* __restrict__ lnw, const float* __restrict__ lnb,
    const _Float16* __restrict__ wLp, const _Float16* __restrict__ wLg,
    const _Float16* __restrict__ wRp, const _Float16* __restrict__ wRg,
    const _Float16* __restrict__ wG,
    const float* __restrict__ bLp, const float* __restrict__ bLg,
    const float* __restrict__ bRp, const float* __restrict__ bRg,
    const float* __restrict__ bG,
    _Float16* __restrict__ leftO, _Float16* __restrict__ rightO,
    float* __restrict__ gO)
{
  constexpr int ALD = C_ + 8;   // f16 row stride (272B, 16B-aligned, conflict-free)
  constexpr int GLD = C_ + 4;   // f32 row stride (528B, 16B-aligned)
  __shared__ _Float16 aln[64 * ALD];
  __shared__ float    gbuf[64 * GLD];

  const int p0 = blockIdx.x * 64;
  const int t  = threadIdx.x;

  // ---- LayerNorm (fp32): 4 threads per position, shfl-quad reduction ----
  {
    const int pos = t >> 2, q = t & 3;
    const v4f* ap = reinterpret_cast<const v4f*>(act + (size_t)(p0 + pos) * C_ + q * 32);
    float v[32];
    float s = 0.f, s2 = 0.f;
#pragma unroll
    for (int i = 0; i < 8; ++i) {
      v4f x = __builtin_nontemporal_load(ap + i);   // act is single-use: NT
      v[4*i] = x.x; v[4*i+1] = x.y; v[4*i+2] = x.z; v[4*i+3] = x.w;
      s  += x.x + x.y + x.z + x.w;
      s2 += x.x*x.x + x.y*x.y + x.z*x.z + x.w*x.w;
    }
    s  += __shfl_xor(s, 1);  s  += __shfl_xor(s, 2);
    s2 += __shfl_xor(s2, 1); s2 += __shfl_xor(s2, 2);
    float mu = s * (1.0f / C_);
    float rs = rsqrtf(s2 * (1.0f / C_) - mu * mu + 1e-5f);
    _Float16* dst = aln + pos * ALD + q * 32;
#pragma unroll
    for (int i = 0; i < 32; ++i)
      dst[i] = (_Float16)((v[i] - mu) * rs * lnw[q*32 + i] + lnb[q*32 + i]);
  }
  __syncthreads();

  const int w    = t >> 5, lane = t & 31;
  const int lrow = lane & 15, ksel = lane >> 4;
  const int pt   = w & 3, h = w >> 2;
  const _Float16* aRow = aln + (pt*16 + lrow) * ALD;
  const int prow = p0 + pt * 16;
  const int mb   = 8 * ksel;     // this lane's first accumulator row (position)

  const v8f vzero = {0.f,0.f,0.f,0.f,0.f,0.f,0.f,0.f};

  // A fragments (activations) loaded once, reused for all 5 weight matrices.
  v16h afr[4];
#pragma unroll
  for (int kc = 0; kc < 4; ++kc) afr[kc] = frag_ld(aRow, kc * 32 + 8 * ksel);

  const _Float16* WP[2]  = {wLp, wRp};
  const _Float16* WG[2]  = {wLg, wRg};
  const float*    BPb[2] = {bLp, bRp};
  const float*    BGb[2] = {bLg, bRg};
  _Float16*       OPo[2] = {leftO, rightO};

  float4 mk0 = *reinterpret_cast<const float4*>(mask + prow + mb);
  float4 mk1 = *reinterpret_cast<const float4*>(mask + prow + mb + 4);
  const float mk[8] = {mk0.x, mk0.y, mk0.z, mk0.w, mk1.x, mk1.y, mk1.z, mk1.w};

  // ---- left pair (pr=0) and right pair (pr=1): proj + gate ----
#pragma unroll
  for (int pr = 0; pr < 2; ++pr) {
    v8f accP[4] = {vzero, vzero, vzero, vzero};
    v8f accG[4] = {vzero, vzero, vzero, vzero};
#pragma unroll
    for (int kc = 0; kc < 4; ++kc) {
      const int k0 = kc * 32 + 8 * ksel;
#pragma unroll
      for (int nt = 0; nt < 4; ++nt) {
        const int crow = h*64 + nt*16 + lrow;           // out-channel row of W
        v16h bp = frag_ld(WP[pr] + crow * C_, k0);
        v16h bg = frag_ld(WG[pr] + crow * C_, k0);
        accP[nt] = WMMA_F16(afr[kc], bp, accP[nt]);
        accG[nt] = WMMA_F16(afr[kc], bg, accG[nt]);
      }
    }
#pragma unroll
    for (int nt = 0; nt < 4; ++nt) {
      const int cc = h*64 + nt*16 + lrow;               // this lane's channel
      const float bp = BPb[pr][cc];
      const float bg = BGb[pr][cc];
      H8 o;
#pragma unroll
      for (int r = 0; r < 8; ++r) {
        float pv = accP[nt][r] + bp;
        float gv = sigmoidf_(accG[nt][r] + bg);
        o.h[r] = (_Float16)(mk[r] * pv * gv);
      }
      // [c][p] layout: lane writes 8 consecutive positions at fixed channel
      *reinterpret_cast<float4*>(OPo[pr] + (size_t)cc * ((size_t)N_ * N_) + prow + mb) = o.q;
    }
  }

  // ---- final gating sigmoid(a @ gating_w.T + b) -> gbuf -> [p][c] global ----
  {
    v8f acc[4] = {vzero, vzero, vzero, vzero};
#pragma unroll
    for (int kc = 0; kc < 4; ++kc) {
      const int k0 = kc * 32 + 8 * ksel;
#pragma unroll
      for (int nt = 0; nt < 4; ++nt) {
        const int crow = h*64 + nt*16 + lrow;
        v16h b = frag_ld(wG + crow * C_, k0);
        acc[nt] = WMMA_F16(afr[kc], b, acc[nt]);
      }
    }
#pragma unroll
    for (int nt = 0; nt < 4; ++nt) {
      const int cc = h*64 + nt*16 + lrow;
      const float bg = bG[cc];
#pragma unroll
      for (int r = 0; r < 8; ++r)
        gbuf[(pt*16 + mb + r) * GLD + cc] = sigmoidf_(acc[nt][r] + bg);
    }
  }
  __syncthreads();
  {
    const int pos = t >> 2, q = t & 3;
    v4f* gp = reinterpret_cast<v4f*>(gO + (size_t)(p0 + pos) * C_ + q * 32);
    const v4f* src = reinterpret_cast<const v4f*>(gbuf + pos * GLD + q * 32);
#pragma unroll
    for (int i = 0; i < 8; ++i)
      __builtin_nontemporal_store(src[i], gp + i);      // g re-read only in k3
  }
}

// ---------------------------------------------------------------------------
// k2: batched per-channel GEMM  t[c][i][j] = sum_k left[c][i][k]*right[c][j][k]
//   Computed as D[m=j][n=i] so each lane's 8 rows are 8 consecutive j
//   (contiguous 32B stores into t[c][i][j]).
//   block tile: 64(j) x 128(i); K staged 64-wide, double-buffered in LDS via
//   GLOBAL_LOAD_ASYNC_TO_LDS_B128 (ASYNCcnt) when available.
// ---------------------------------------------------------------------------
__global__ __launch_bounds__(256) void k2_gemm(
    const _Float16* __restrict__ leftB, const _Float16* __restrict__ rightB,
    float* __restrict__ tb)
{
  constexpr int LD = 72;                       // 64 + 8 pad (144B rows)
  __shared__ _Float16 at[2][64  * LD];         // right rows (j)
  __shared__ _Float16 bt[2][128 * LD];         // left  rows (i)
  const int c  = blockIdx.z;
  const int j0 = blockIdx.x * 64;
  const int i0 = blockIdx.y * 128;
  const _Float16* Rp = rightB + (size_t)c * N_ * N_;
  const _Float16* Lp = leftB  + (size_t)c * N_ * N_;
  const int t = threadIdx.x, w = t >> 5, lane = t & 31;
  const int wm = w & 1, wn = w >> 1;           // wave grid: 2 (j) x 4 (i)
  const int lrow = lane & 15, ksel = lane >> 4;

  auto load_stage = [&](int buf, int k0g) {
#pragma unroll
    for (int s = 0; s < 2; ++s) {              // A tile: 64 rows x 128B
      int ch = t + 256 * s;
      int r = ch >> 3, col = (ch & 7) * 8;
      cp16_to_lds(at[buf] + r * LD + col, Rp + (size_t)(j0 + r) * N_ + k0g + col);
    }
#pragma unroll
    for (int s = 0; s < 4; ++s) {              // B tile: 128 rows x 128B
      int ch = t + 256 * s;
      int r = ch >> 3, col = (ch & 7) * 8;
      cp16_to_lds(bt[buf] + r * LD + col, Lp + (size_t)(i0 + r) * N_ + k0g + col);
    }
  };

  const v8f vzero = {0.f,0.f,0.f,0.f,0.f,0.f,0.f,0.f};
  v8f acc[2][2] = {{vzero, vzero}, {vzero, vzero}};

  load_stage(0, 0);
  constexpr int KT = N_ / 64;                  // 8 K-steps
  for (int kt = 0; kt < KT; ++kt) {
    wait_cp_async();                           // my async loads for buf kt&1 done
    __syncthreads();                           // everyone's are done
    if (kt + 1 < KT) load_stage((kt + 1) & 1, (kt + 1) * 64);
    const _Float16* A = at[kt & 1];
    const _Float16* B = bt[kt & 1];
#pragma unroll
    for (int kk = 0; kk < 2; ++kk) {
      const int k0 = kk * 32 + 8 * ksel;
      v16h a0 = frag_ld(A + (wm*32 +      lrow) * LD, k0);
      v16h a1 = frag_ld(A + (wm*32 + 16 + lrow) * LD, k0);
      v16h b0 = frag_ld(B + (wn*32 +      lrow) * LD, k0);
      v16h b1 = frag_ld(B + (wn*32 + 16 + lrow) * LD, k0);
      acc[0][0] = WMMA_F16(a0, b0, acc[0][0]);
      acc[0][1] = WMMA_F16(a0, b1, acc[0][1]);
      acc[1][0] = WMMA_F16(a1, b0, acc[1][0]);
      acc[1][1] = WMMA_F16(a1, b1, acc[1][1]);
    }
  }
#pragma unroll
  for (int mi = 0; mi < 2; ++mi) {
#pragma unroll
    for (int ni = 0; ni < 2; ++ni) {
      const int i = i0 + wn*32 + ni*16 + lrow;       // fixed i per lane
      const int j = j0 + wm*32 + mi*16 + 8 * ksel;   // 8 consecutive j
      F8 f; f.v = acc[mi][ni];
      v4f* dst = reinterpret_cast<v4f*>(tb + ((size_t)c * N_ + i) * N_ + j);
      __builtin_nontemporal_store(f.p[0], dst);      // t read once in k3: NT,
      __builtin_nontemporal_store(f.p[1], dst + 1);  // keep left/right hot in L2
    }
  }
}

// ---------------------------------------------------------------------------
// k3: LDS transpose of t tile -> LayerNorm over c -> out projection as
//   D[m=c][n=p] (A = out_proj_w rows, B = normalized tile) -> *gate -> out[p][c]
// ---------------------------------------------------------------------------
__global__ __launch_bounds__(256) void k3_final(
    const float* __restrict__ tb, const float* __restrict__ gO,
    const float* __restrict__ lncw, const float* __restrict__ lncb,
    const _Float16* __restrict__ wO, const float* __restrict__ bO,
    float* __restrict__ out)
{
  constexpr int TLD = C_ + 1;   // 129 (conflict-free transpose writes)
  constexpr int ALD = C_ + 8;   // 136
  __shared__ float    tt [64 * TLD];
  __shared__ _Float16 tln[64 * ALD];
  const int p0 = blockIdx.x * 64;
  const int t  = threadIdx.x;

  // transposed tile load: t[c][p] -> tt[p][c]
  {
    const int c = t >> 1, pc = (t & 1) * 32;
    const v4f* src = reinterpret_cast<const v4f*>(tb + (size_t)c * N_ * N_ + p0 + pc);
#pragma unroll
    for (int s = 0; s < 8; ++s) {
      v4f v = __builtin_nontemporal_load(src + s);
      tt[(pc + 4*s + 0) * TLD + c] = v.x;
      tt[(pc + 4*s + 1) * TLD + c] = v.y;
      tt[(pc + 4*s + 2) * TLD + c] = v.z;
      tt[(pc + 4*s + 3) * TLD + c] = v.w;
    }
  }
  __syncthreads();
  // LayerNorm over c (fp32)
  {
    const int pos = t >> 2, q = t & 3;
    const float* row = tt + pos * TLD + q * 32;
    float s = 0.f, s2 = 0.f;
#pragma unroll
    for (int i = 0; i < 32; ++i) { float x = row[i]; s += x; s2 += x * x; }
    s  += __shfl_xor(s, 1);  s  += __shfl_xor(s, 2);
    s2 += __shfl_xor(s2, 1); s2 += __shfl_xor(s2, 2);
    float mu = s * (1.0f / C_);
    float rs = rsqrtf(s2 * (1.0f / C_) - mu * mu + 1e-5f);
    _Float16* dst = tln + pos * ALD + q * 32;
#pragma unroll
    for (int i = 0; i < 32; ++i)
      dst[i] = (_Float16)((row[i] - mu) * rs * lncw[q*32 + i] + lncb[q*32 + i]);
  }
  __syncthreads();

  const int w = t >> 5, lane = t & 31;
  const int lrow = lane & 15, ksel = lane >> 4;
  const int pt = w & 3, h = w >> 2;
  const _Float16* bRow = tln + (pt*16 + lrow) * ALD;  // B: column n = position
  const v8f vzero = {0.f,0.f,0.f,0.f,0.f,0.f,0.f,0.f};
  v8f acc[4] = {vzero, vzero, vzero, vzero};
#pragma unroll
  for (int kc = 0; kc < 4; ++kc) {
    const int k0 = kc * 32 + 8 * ksel;
    v16h b = frag_ld(bRow, k0);
#pragma unroll
    for (int mt = 0; mt < 4; ++mt) {
      const int crow = h*64 + mt*16 + lrow;           // A: row m = out channel
      v16h a = frag_ld(wO + crow * C_, k0);
      acc[mt] = WMMA_F16(a, b, acc[mt]);
    }
  }
  const int p = p0 + pt*16 + lrow;                    // fixed position per lane
  float* op = out + (size_t)p * C_;
  const float* gp = gO + (size_t)p * C_;
#pragma unroll
  for (int mt = 0; mt < 4; ++mt) {
    const int cb = h*64 + mt*16 + 8 * ksel;           // 8 consecutive channels
    v4f g0 = __builtin_nontemporal_load(reinterpret_cast<const v4f*>(gp + cb));
    v4f g1 = __builtin_nontemporal_load(reinterpret_cast<const v4f*>(gp + cb + 4));
    float4 b0 = *reinterpret_cast<const float4*>(bO + cb);
    float4 b1 = *reinterpret_cast<const float4*>(bO + cb + 4);
    F8 f; f.v = acc[mt];
    v4f o0 = {(f.f[0] + b0.x) * g0.x, (f.f[1] + b0.y) * g0.y,
              (f.f[2] + b0.z) * g0.z, (f.f[3] + b0.w) * g0.w};
    v4f o1 = {(f.f[4] + b1.x) * g1.x, (f.f[5] + b1.y) * g1.y,
              (f.f[6] + b1.z) * g1.z, (f.f[7] + b1.w) * g1.w};
    __builtin_nontemporal_store(o0, reinterpret_cast<v4f*>(op + cb));
    __builtin_nontemporal_store(o1, reinterpret_cast<v4f*>(op + cb + 4));
  }
}

// ---------------------------------------------------------------------------
extern "C" void kernel_launch(void* const* d_in, const int* in_sizes, int n_in,
                              void* d_out, int out_size, void* d_ws, size_t ws_size,
                              hipStream_t stream)
{
  const float* act   = (const float*)d_in[0];
  const float* mask  = (const float*)d_in[1];
  const float* lnw   = (const float*)d_in[2];
  const float* lnb   = (const float*)d_in[3];
  const float* wLp_f = (const float*)d_in[4];
  const float* bLp   = (const float*)d_in[5];
  const float* wRp_f = (const float*)d_in[6];
  const float* bRp   = (const float*)d_in[7];
  const float* wLg_f = (const float*)d_in[8];
  const float* bLg   = (const float*)d_in[9];
  const float* wRg_f = (const float*)d_in[10];
  const float* bRg   = (const float*)d_in[11];
  const float* lncw  = (const float*)d_in[12];
  const float* lncb  = (const float*)d_in[13];
  const float* wO_f  = (const float*)d_in[14];
  const float* bO    = (const float*)d_in[15];
  const float* wG_f  = (const float*)d_in[16];
  const float* bG    = (const float*)d_in[17];

  char* ws = (char*)d_ws;
  const size_t WE    = (size_t)C_ * C_;       // 16384 elems per weight matrix
  const size_t PLANE = (size_t)N_ * N_;       // 262144 positions

  _Float16* wLp = (_Float16*)ws;
  _Float16* wLg = wLp + WE;
  _Float16* wRp = wLg + WE;
  _Float16* wRg = wRp + WE;
  _Float16* wG  = wRg + WE;
  _Float16* wO  = wG  + WE;

  _Float16* leftB  = (_Float16*)(ws + (size_t)262144);              // [C][N][N] f16
  _Float16* rightB = leftB + (size_t)C_ * PLANE;                    // [C][N][N] f16
  float*    g      = (float*)(ws + (size_t)262144 +
                              2 * (size_t)C_ * PLANE * sizeof(_Float16)); // [N*N][C] f32
  float*    tb     = g + (size_t)C_ * PLANE;                        // [C][N][N] f32

  k0_cvt_w<<<64, 256, 0, stream>>>(wLp_f, wLp);
  k0_cvt_w<<<64, 256, 0, stream>>>(wLg_f, wLg);
  k0_cvt_w<<<64, 256, 0, stream>>>(wRp_f, wRp);
  k0_cvt_w<<<64, 256, 0, stream>>>(wRg_f, wRg);
  k0_cvt_w<<<64, 256, 0, stream>>>(wG_f,  wG);
  k0_cvt_w<<<64, 256, 0, stream>>>(wO_f,  wO);

  k1_prep<<<(N_ * N_) / 64, 256, 0, stream>>>(
      act, mask, lnw, lnb, wLp, wLg, wRp, wRg, wG,
      bLp, bLg, bRp, bRg, bG, leftB, rightB, g);

  dim3 g2(N_ / 64, N_ / 128, C_);
  k2_gemm<<<g2, 256, 0, stream>>>(leftB, rightB, tb);

  k3_final<<<(N_ * N_) / 64, 256, 0, stream>>>(tb, g, lncw, lncb, wO, bO,
                                               (float*)d_out);
}
